// Self_Attention_Global_Block_11141145166182
// MI455X (gfx1250) — compile-verified, matrix operand order checked by reading, not run on hardware
//
#include <hip/hip_runtime.h>
#include <math.h>

// MI455X / gfx1250, wave32. All GEMMs via v_wmma_f32_16x16x32_bf16.
// Effective problem (after dropping the duplicated batch): B=32, C=512, N=256, H=8.
// GEMM core: 128x128 block tile, 8 waves (4Mx2N), 32x64 per wave = 8 WMMA / K-step,
// double-buffered LDS -> 1 barrier per K-step.

typedef __attribute__((ext_vector_type(16))) __bf16 v16bf;
typedef __attribute__((ext_vector_type(8)))  float  v8f;

__device__ __forceinline__ unsigned short f2b(float f) {
  unsigned int u = __float_as_uint(f);
  u += 0x7FFFu + ((u >> 16) & 1u);           // round-to-nearest-even
  return (unsigned short)(u >> 16);
}
__device__ __forceinline__ float b2f(unsigned short h) {
  return __uint_as_float(((unsigned int)h) << 16);
}

union FragU { v16bf v; unsigned int u[8]; };

#define LDSS 40                 // padded LDS row stride (bf16) -> 80B rows, conflict-free
#define LBUF (128 * LDSS)       // one LDS buffer (ushorts)

// ---------------------------------------------------------------------------
// C(128x128,f32) += A(MxK bf16 row-major) * Bt^T, Bt stored [n][k].
// Fragment swizzle per CDNA5 ISA 7.12.2 (16-bit A 16x32): lane g=lane>>4,
// dword p holds K = ((p&4)<<2) + 8g + 2(p&3) (+1). Dwords p=0..3 / 4..7 are
// byte-contiguous -> compiler merges to ds_load_b128 pairs.
// ---------------------------------------------------------------------------
__device__ __forceinline__ void gemm128(
    const unsigned short* __restrict__ A, int lda,
    const unsigned short* __restrict__ Bt, int ldb,
    int m0, int n0, int K,
    unsigned short* shA, unsigned short* shB,   // each 2*LBUF
    v8f acc[2][4])
{
  const int t    = threadIdx.x;
  const int r0s  = t >> 2;            // staging rows 0..63   (chunk t)
  const int r1s  = r0s + 64;          // staging rows 64..127 (chunk t+256)
  const int kqs  = (t & 3) << 3;      // 8 bf16 = 16B
  const int wave = t >> 5;
  const int lane = t & 31;
  const int wm   = wave >> 1;         // 0..3 -> M offset 32*wm
  const int wn   = wave & 1;          // 0..1 -> N offset 64*wn
  const int fm   = lane & 15;
  const int g    = lane >> 4;

  #pragma unroll
  for (int ai = 0; ai < 2; ++ai)
    #pragma unroll
    for (int bi = 0; bi < 4; ++bi)
      #pragma unroll
      for (int i = 0; i < 8; ++i) acc[ai][bi][i] = 0.0f;

  const unsigned short* aP0 = A  + (size_t)(m0 + r0s) * lda + kqs;
  const unsigned short* aP1 = A  + (size_t)(m0 + r1s) * lda + kqs;
  const unsigned short* bP0 = Bt + (size_t)(n0 + r0s) * ldb + kqs;
  const unsigned short* bP1 = Bt + (size_t)(n0 + r1s) * ldb + kqs;
  const int sO0 = r0s * LDSS + kqs;
  const int sO1 = r1s * LDSS + kqs;

  // prologue: stage K-tile 0 into buffer 0
  {
    uint4 a0 = *(const uint4*)(aP0);
    uint4 a1 = *(const uint4*)(aP1);
    uint4 b0 = *(const uint4*)(bP0);
    uint4 b1 = *(const uint4*)(bP1);
    *(uint4*)(shA + sO0) = a0;  *(uint4*)(shA + sO1) = a1;
    *(uint4*)(shB + sO0) = b0;  *(uint4*)(shB + sO1) = b1;
  }
  __syncthreads();

  int cur = 0;
  for (int k0 = 0; k0 < K; k0 += 32) {
    const int nxt = cur ^ 1;
    if (k0 + 32 < K) {                       // stage next tile into other buffer
      uint4 a0 = *(const uint4*)(aP0 + k0 + 32);
      uint4 a1 = *(const uint4*)(aP1 + k0 + 32);
      uint4 b0 = *(const uint4*)(bP0 + k0 + 32);
      uint4 b1 = *(const uint4*)(bP1 + k0 + 32);
      *(uint4*)(shA + nxt * LBUF + sO0) = a0;
      *(uint4*)(shA + nxt * LBUF + sO1) = a1;
      *(uint4*)(shB + nxt * LBUF + sO0) = b0;
      *(uint4*)(shB + nxt * LBUF + sO1) = b1;
      if (k0 + 64 < K) {                     // global_prefetch_b8 two tiles ahead
        __builtin_prefetch(aP0 + k0 + 64, 0, 1);
        __builtin_prefetch(aP1 + k0 + 64, 0, 1);
        __builtin_prefetch(bP0 + k0 + 64, 0, 1);
        __builtin_prefetch(bP1 + k0 + 64, 0, 1);
      }
    }
    const unsigned short* sa = shA + cur * LBUF;
    const unsigned short* sb = shB + cur * LBUF;
    FragU fa[2], fb[4];
    #pragma unroll
    for (int p = 0; p < 8; ++p) {
      const int kk = ((p & 4) << 2) + (g << 3) + ((p & 3) << 1);
      fa[0].u[p] = *(const unsigned int*)(sa + (wm * 32 + fm) * LDSS + kk);
      fa[1].u[p] = *(const unsigned int*)(sa + (wm * 32 + 16 + fm) * LDSS + kk);
      fb[0].u[p] = *(const unsigned int*)(sb + (wn * 64 + fm) * LDSS + kk);
      fb[1].u[p] = *(const unsigned int*)(sb + (wn * 64 + 16 + fm) * LDSS + kk);
      fb[2].u[p] = *(const unsigned int*)(sb + (wn * 64 + 32 + fm) * LDSS + kk);
      fb[3].u[p] = *(const unsigned int*)(sb + (wn * 64 + 48 + fm) * LDSS + kk);
    }
    #pragma unroll
    for (int ai = 0; ai < 2; ++ai)
      #pragma unroll
      for (int bi = 0; bi < 4; ++bi)
        acc[ai][bi] = __builtin_amdgcn_wmma_f32_16x16x32_bf16(
            false, fa[ai].v, false, fb[bi].v, (short)0, acc[ai][bi], false, false);
    __syncthreads();   // next-buffer writes visible; cur reads done before overwrite
    cur = nxt;
  }
}

// Output coords of acc[ai][bi]: 8 consecutive rows starting r, column c.
__device__ __forceinline__ void tile_coords(int m0, int n0, int ai, int bi, int& r, int& c) {
  const int wave = threadIdx.x >> 5, lane = threadIdx.x & 31;
  r = m0 + (wave >> 1) * 32 + ai * 16 + ((lane >> 4) << 3);
  c = n0 + (wave & 1) * 64 + bi * 16 + (lane & 15);
}

// ---------------------------------------------------------------------------
// K0: fp32 -> bf16 conversions (weights pre-transposed so B operands are [n][k])
// ---------------------------------------------------------------------------
__global__ void k_convert(const float* __restrict__ x, const float* __restrict__ proj_w,
                          const float* __restrict__ wq, const float* __restrict__ wk,
                          const float* __restrict__ wv, const float* __restrict__ wo,
                          const float* __restrict__ mo_w,
                          unsigned short* xb, unsigned short* pwb,
                          unsigned short* wqT, unsigned short* wkT, unsigned short* wvT,
                          unsigned short* woT, unsigned short* mowb)
{
  const size_t S0 = 32u * 512u * 256u;   // xb: x is already [b][c][dhw]
  const size_t S1 = 256u * 256u;         // proj_w [n][k]
  const size_t S2 = 512u * 4096u;        // each of wq/wk/wv -> [cout][c]
  const size_t S5 = 4096u * 512u;        // wo -> [c][j]
  const size_t S6 = 512u * 512u;         // mo_w [o][c]
  const size_t total = S0 + S1 + 3 * S2 + S5 + S6;
  const size_t stride = (size_t)gridDim.x * blockDim.x;
  for (size_t i = (size_t)blockIdx.x * blockDim.x + threadIdx.x; i < total; i += stride) {
    size_t idx = i;
    if (idx < S0) { xb[idx] = f2b(x[idx]); continue; }
    idx -= S0;
    if (idx < S1) { pwb[idx] = f2b(proj_w[idx]); continue; }
    idx -= S1;
    if (idx < S2) { size_t co = idx / 512, c = idx % 512; wqT[idx] = f2b(wq[c * 4096 + co]); continue; }
    idx -= S2;
    if (idx < S2) { size_t co = idx / 512, c = idx % 512; wkT[idx] = f2b(wk[c * 4096 + co]); continue; }
    idx -= S2;
    if (idx < S2) { size_t co = idx / 512, c = idx % 512; wvT[idx] = f2b(wv[c * 4096 + co]); continue; }
    idx -= S2;
    if (idx < S5) { size_t c = idx / 4096, j = idx % 4096; woT[idx] = f2b(wo[j * 512 + c]); continue; }
    idx -= S5;
    mowb[idx] = f2b(mo_w[idx]);
  }
}

// ---------------------------------------------------------------------------
// K1: patch embed. emb[b][n][c] = sum_k proj_w[n][k] * x[b][c][k] + proj_b[n]
// ---------------------------------------------------------------------------
__global__ void k_embed(const unsigned short* __restrict__ pwb,
                        const unsigned short* __restrict__ xb,
                        const float* __restrict__ proj_b, float* __restrict__ emb)
{
  __shared__ unsigned short shA[2 * LBUF], shB[2 * LBUF];
  const int b  = blockIdx.y;
  const int m0 = (blockIdx.x >> 2) * 128;   // n (seq), 2 tiles
  const int n0 = (blockIdx.x & 3) * 128;    // c, 4 tiles
  v8f acc[2][4];
  gemm128(pwb, 256, xb + (size_t)b * 512 * 256, 256, m0, n0, 256, shA, shB, acc);
  float* ob = emb + (size_t)b * 256 * 512;
  #pragma unroll
  for (int ai = 0; ai < 2; ++ai)
    #pragma unroll
    for (int bi = 0; bi < 4; ++bi) {
      int r0, c0; tile_coords(m0, n0, ai, bi, r0, c0);
      #pragma unroll
      for (int r = 0; r < 8; ++r) {
        const int n = r0 + r;
        ob[(size_t)n * 512 + c0] = acc[ai][bi][r] + proj_b[n];
      }
    }
}

// ---------------------------------------------------------------------------
// LayerNorm over C=512 per row (ln1 & ln2), writes bf16.
// ---------------------------------------------------------------------------
__global__ void k_layernorm(const float* __restrict__ in, const float* __restrict__ gamma,
                            const float* __restrict__ beta, unsigned short* __restrict__ out)
{
  __shared__ float red[256];
  const size_t row = blockIdx.x;
  const float* p = in + row * 512;
  const int t = threadIdx.x;
  float x0 = p[t], x1 = p[t + 256];
  red[t] = x0 + x1; __syncthreads();
  for (int o = 128; o > 0; o >>= 1) { if (t < o) red[t] += red[t + o]; __syncthreads(); }
  const float mu = red[0] * (1.0f / 512.0f); __syncthreads();
  const float d0 = x0 - mu, d1 = x1 - mu;
  red[t] = d0 * d0 + d1 * d1; __syncthreads();
  for (int o = 128; o > 0; o >>= 1) { if (t < o) red[t] += red[t + o]; __syncthreads(); }
  const float rs = rsqrtf(red[0] * (1.0f / 512.0f) + 1e-6f);
  unsigned short* q = out + row * 512;
  q[t]       = f2b(d0 * rs * gamma[t] + beta[t]);
  q[t + 256] = f2b(d1 * rs * gamma[t + 256] + beta[t + 256]);
}

// ---------------------------------------------------------------------------
// QKV per (b,h). grid.z: 0=q,1=k (write [bh][i][n] transposed), 2=v ([bh][n][i]).
// ---------------------------------------------------------------------------
__global__ void k_qkv(const unsigned short* __restrict__ xnb,
                      const unsigned short* __restrict__ wqT,
                      const unsigned short* __restrict__ wkT,
                      const unsigned short* __restrict__ wvT,
                      unsigned short* __restrict__ qT, unsigned short* __restrict__ kT,
                      unsigned short* __restrict__ vC, int b0)
{
  __shared__ unsigned short shA[2 * LBUF], shB[2 * LBUF];
  const int bh = blockIdx.y;
  const int b = b0 + (bh >> 3), h = bh & 7;
  const int mode = blockIdx.z;
  const unsigned short* wT = (mode == 0) ? wqT : ((mode == 1) ? wkT : wvT);
  const int m0 = (blockIdx.x >> 2) * 128;   // n (seq), 2 tiles
  const int n0 = (blockIdx.x & 3) * 128;    // i (within head), 4 tiles
  v8f acc[2][4];
  gemm128(xnb + (size_t)b * 256 * 512, 512, wT + (size_t)h * 512 * 512, 512,
          m0, n0, 512, shA, shB, acc);
  if (mode < 2) {
    unsigned short* dst = ((mode == 0) ? qT : kT) + (size_t)bh * 512 * 256;
    #pragma unroll
    for (int ai = 0; ai < 2; ++ai)
      #pragma unroll
      for (int bi = 0; bi < 4; ++bi) {
        int r0, c0; tile_coords(m0, n0, ai, bi, r0, c0);
        union { unsigned short s[8]; uint4 q; } pk;
        #pragma unroll
        for (int r = 0; r < 8; ++r) pk.s[r] = f2b(acc[ai][bi][r]);
        *(uint4*)&dst[(size_t)c0 * 256 + r0] = pk.q;   // contiguous in n
      }
  } else {
    unsigned short* dst = vC + (size_t)bh * 256 * 512;
    #pragma unroll
    for (int ai = 0; ai < 2; ++ai)
      #pragma unroll
      for (int bi = 0; bi < 4; ++bi) {
        int r0, c0; tile_coords(m0, n0, ai, bi, r0, c0);
        #pragma unroll
        for (int r = 0; r < 8; ++r)
          dst[(size_t)(r0 + r) * 512 + c0] = f2b(acc[ai][bi][r]);
      }
  }
}

// ---------------------------------------------------------------------------
// scores[i][j] = sum_n qT[i][n] * kT[j][n]; store bf16 + per-tile sum/sumsq.
// ---------------------------------------------------------------------------
__global__ void k_scores(const unsigned short* __restrict__ qT,
                         const unsigned short* __restrict__ kT,
                         unsigned short* __restrict__ attn, float* __restrict__ part)
{
  __shared__ unsigned short shA[2 * LBUF], shB[2 * LBUF];
  __shared__ float redS[256], redQ[256];
  const int bh = blockIdx.y;
  const int m0 = (blockIdx.x >> 2) * 128;   // i, 4 tiles
  const int n0 = (blockIdx.x & 3) * 128;    // j, 4 tiles
  v8f acc[2][4];
  gemm128(qT + (size_t)bh * 512 * 256, 256, kT + (size_t)bh * 512 * 256, 256,
          m0, n0, 256, shA, shB, acc);
  unsigned short* dst = attn + (size_t)bh * 512 * 512;
  float s = 0.0f, s2 = 0.0f;
  #pragma unroll
  for (int ai = 0; ai < 2; ++ai)
    #pragma unroll
    for (int bi = 0; bi < 4; ++bi) {
      int r0, c0; tile_coords(m0, n0, ai, bi, r0, c0);
      #pragma unroll
      for (int r = 0; r < 8; ++r) {
        const float v = acc[ai][bi][r];
        dst[(size_t)(r0 + r) * 512 + c0] = f2b(v);
        s += v; s2 += v * v;
      }
    }
  const int t = threadIdx.x;
  redS[t] = s; redQ[t] = s2; __syncthreads();
  for (int o = 128; o > 0; o >>= 1) {
    if (t < o) { redS[t] += redS[t + o]; redQ[t] += redQ[t + o]; }
    __syncthreads();
  }
  if (t == 0) {
    part[((size_t)bh * 16 + blockIdx.x) * 2 + 0] = redS[0];
    part[((size_t)bh * 16 + blockIdx.x) * 2 + 1] = redQ[0];
  }
}

__global__ void k_stats(const float* __restrict__ part, float* __restrict__ stats)
{
  __shared__ float rs[16], rq[16];
  const int bh = blockIdx.x, t = threadIdx.x;   // 16 threads
  rs[t] = part[((size_t)bh * 16 + t) * 2 + 0];
  rq[t] = part[((size_t)bh * 16 + t) * 2 + 1];
  __syncthreads();
  for (int o = 8; o > 0; o >>= 1) {
    if (t < o) { rs[t] += rs[t + o]; rq[t] += rq[t + o]; }
    __syncthreads();
  }
  if (t == 0) {
    const float inv = 1.0f / 262144.0f;
    const float mu = rs[0] * inv;
    const float var = rq[0] * inv - mu * mu;
    stats[bh * 2 + 0] = mu;
    stats[bh * 2 + 1] = rsqrtf(var + 1e-5f);   // InstanceNorm2d eps
  }
}

// instance-norm + row softmax, in place (bf16).
__global__ void k_softmax(unsigned short* __restrict__ attn, const float* __restrict__ stats)
{
  __shared__ float red[256];
  const int bh = blockIdx.y, i = blockIdx.x, t = threadIdx.x;
  unsigned short* row = attn + ((size_t)bh * 512 + i) * 512;
  const float mu = stats[bh * 2 + 0], rs = stats[bh * 2 + 1];
  float x0 = (b2f(row[t]) - mu) * rs;
  float x1 = (b2f(row[t + 256]) - mu) * rs;
  red[t] = fmaxf(x0, x1); __syncthreads();
  for (int o = 128; o > 0; o >>= 1) { if (t < o) red[t] = fmaxf(red[t], red[t + o]); __syncthreads(); }
  const float m = red[0]; __syncthreads();
  const float e0 = expf(x0 - m), e1 = expf(x1 - m);
  red[t] = e0 + e1; __syncthreads();
  for (int o = 128; o > 0; o >>= 1) { if (t < o) red[t] += red[t + o]; __syncthreads(); }
  const float inv = 1.0f / red[0];
  row[t]       = f2b(e0 * inv);
  row[t + 256] = f2b(e1 * inv);
}

// ---------------------------------------------------------------------------
// ctx[i][nseq] = sum_j attn[i][j] * v[nseq][j]; scatter to ctx2[bL][n][i*8+h].
// ---------------------------------------------------------------------------
__global__ void k_ctx(const unsigned short* __restrict__ attn,
                      const unsigned short* __restrict__ vC,
                      unsigned short* __restrict__ ctx2)
{
  __shared__ unsigned short shA[2 * LBUF], shB[2 * LBUF];
  const int bh = blockIdx.y;
  const int bL = bh >> 3, h = bh & 7;
  const int m0 = (blockIdx.x >> 1) * 128;   // i, 4 tiles
  const int n0 = (blockIdx.x & 1) * 128;    // nseq, 2 tiles
  v8f acc[2][4];
  gemm128(attn + (size_t)bh * 512 * 512, 512, vC + (size_t)bh * 256 * 512, 512,
          m0, n0, 512, shA, shB, acc);
  unsigned short* dst = ctx2 + (size_t)bL * 256 * 4096;
  #pragma unroll
  for (int ai = 0; ai < 2; ++ai)
    #pragma unroll
    for (int bi = 0; bi < 4; ++bi) {
      int r0, c0; tile_coords(m0, n0, ai, bi, r0, c0);   // rows=i, cols=nseq
      #pragma unroll
      for (int r = 0; r < 8; ++r)
        dst[(size_t)c0 * 4096 + (size_t)(r0 + r) * 8 + h] = f2b(acc[ai][bi][r]);
    }
}

// out[n][c] = ctx2[n][:] @ wo + emb (residual), f32.
__global__ void k_outproj(const unsigned short* __restrict__ ctx2,
                          const unsigned short* __restrict__ woT,
                          const float* __restrict__ emb, float* __restrict__ y1, int b0)
{
  __shared__ unsigned short shA[2 * LBUF], shB[2 * LBUF];
  const int bL = blockIdx.y, b = b0 + bL;
  const int m0 = (blockIdx.x >> 2) * 128;   // n, 2 tiles
  const int n0 = (blockIdx.x & 3) * 128;    // c, 4 tiles
  v8f acc[2][4];
  gemm128(ctx2 + (size_t)bL * 256 * 4096, 4096, woT, 4096, m0, n0, 4096, shA, shB, acc);
  const float* res = emb + (size_t)b * 256 * 512;
  float* dst = y1 + (size_t)b * 256 * 512;
  #pragma unroll
  for (int ai = 0; ai < 2; ++ai)
    #pragma unroll
    for (int bi = 0; bi < 4; ++bi) {
      int r0, c0; tile_coords(m0, n0, ai, bi, r0, c0);
      #pragma unroll
      for (int r = 0; r < 8; ++r) {
        const size_t o = (size_t)(r0 + r) * 512 + c0;
        dst[o] = acc[ai][bi][r] + res[o];
      }
    }
}

// final[b][o][pos] = gelu( y2[b][pos][:] @ mo_w[o][:] + mo_b[o] ), exact erf GELU.
__global__ void k_mo(const unsigned short* __restrict__ y2,
                     const unsigned short* __restrict__ mowb,
                     const float* __restrict__ mo_b, float* __restrict__ out)
{
  __shared__ unsigned short shA[2 * LBUF], shB[2 * LBUF];
  const int b = blockIdx.y;
  const int m0 = (blockIdx.x >> 2) * 128;   // pos, 2 tiles
  const int n0 = (blockIdx.x & 3) * 128;    // o, 4 tiles
  v8f acc[2][4];
  gemm128(y2 + (size_t)b * 256 * 512, 512, mowb, 512, m0, n0, 512, shA, shB, acc);
  #pragma unroll
  for (int ai = 0; ai < 2; ++ai)
    #pragma unroll
    for (int bi = 0; bi < 4; ++bi) {
      int r0, c0; tile_coords(m0, n0, ai, bi, r0, c0);  // rows=pos (8 contiguous), cols=o
      const float bias = mo_b[c0];
      union { float f[8]; float4 q[2]; } pk;
      #pragma unroll
      for (int r = 0; r < 8; ++r) {
        const float tv = acc[ai][bi][r] + bias;
        pk.f[r] = 0.5f * tv * (1.0f + erff(tv * 0.70710678118654752f));
      }
      float* dst = out + ((size_t)b * 512 + c0) * 256 + r0;
      *(float4*)(dst + 0) = pk.q[0];
      *(float4*)(dst + 4) = pk.q[1];
    }
}

// ---------------------------------------------------------------------------
extern "C" void kernel_launch(void* const* d_in, const int* in_sizes, int n_in,
                              void* d_out, int out_size, void* d_ws, size_t ws_size,
                              hipStream_t stream)
{
  (void)in_sizes; (void)n_in; (void)out_size; (void)ws_size;
  const float* x      = (const float*)d_in[0];
  const float* proj_w = (const float*)d_in[1];
  const float* proj_b = (const float*)d_in[2];
  const float* ln1_g  = (const float*)d_in[3];
  const float* ln1_b  = (const float*)d_in[4];
  const float* wq     = (const float*)d_in[5];
  const float* wk     = (const float*)d_in[6];
  const float* wv     = (const float*)d_in[7];
  const float* wo     = (const float*)d_in[8];
  const float* ln2_g  = (const float*)d_in[9];
  const float* ln2_b  = (const float*)d_in[10];
  const float* mo_w   = (const float*)d_in[11];
  const float* mo_b   = (const float*)d_in[12];
  float* outp = (float*)d_out;

  char* w = (char*)d_ws;
  size_t off = 0;
  auto nxt = [&](size_t bytes) { char* p = w + off; off = (off + bytes + 255) & ~(size_t)255; return p; };
  unsigned short* xb    = (unsigned short*)nxt(8388608);    // x bf16 [b][c][k]
  unsigned short* pwb   = (unsigned short*)nxt(131072);     // proj_w bf16
  unsigned short* wqT   = (unsigned short*)nxt(4194304);    // [cout][c]
  unsigned short* wkT   = (unsigned short*)nxt(4194304);
  unsigned short* wvT   = (unsigned short*)nxt(4194304);
  unsigned short* woT   = (unsigned short*)nxt(4194304);    // [c][j]
  unsigned short* mowb  = (unsigned short*)nxt(524288);     // [o][c]
  float*          emb   = (float*)nxt(16777216);            // residual, f32
  unsigned short* xnb   = (unsigned short*)nxt(8388608);    // LN1(emb) bf16
  unsigned short* qT    = (unsigned short*)nxt(16777216);   // chunk: 64 bh x 512 x 256
  unsigned short* kT    = (unsigned short*)nxt(16777216);
  unsigned short* vC    = (unsigned short*)nxt(16777216);   // chunk: 64 bh x 256 x 512
  unsigned short* attn  = (unsigned short*)nxt(33554432);   // chunk: 64 bh x 512 x 512
  float*          part  = (float*)nxt(8192);                // 64 bh x 16 tiles x {sum,sumsq}
  float*          stats = (float*)nxt(512);                 // 64 bh x {mu, rsig}
  unsigned short* ctx2  = (unsigned short*)nxt(16777216);   // chunk: 8 b x 256 x 4096
  float*          y1    = (float*)nxt(16777216);            // attn out + residual, f32
  unsigned short* y2    = (unsigned short*)nxt(8388608);    // LN2 bf16

  k_convert<<<4096, 256, 0, stream>>>(x, proj_w, wq, wk, wv, wo, mo_w,
                                      xb, pwb, wqT, wkT, wvT, woT, mowb);
  k_embed<<<dim3(8, 32), 256, 0, stream>>>(pwb, xb, proj_b, emb);
  k_layernorm<<<8192, 256, 0, stream>>>(emb, ln1_g, ln1_b, xnb);

  for (int pass = 0; pass < 4; ++pass) {           // 8 batches (64 bh) per pass
    const int b0 = pass * 8;
    k_qkv<<<dim3(8, 64, 3), 256, 0, stream>>>(xnb, wqT, wkT, wvT, qT, kT, vC, b0);
    k_scores<<<dim3(16, 64), 256, 0, stream>>>(qT, kT, attn, part);
    k_stats<<<64, 16, 0, stream>>>(part, stats);
    k_softmax<<<dim3(512, 64), 256, 0, stream>>>(attn, stats);
    k_ctx<<<dim3(8, 64), 256, 0, stream>>>(attn, vC, ctx2);
    k_outproj<<<dim3(8, 8), 256, 0, stream>>>(ctx2, woT, emb, y1, b0);
  }

  k_layernorm<<<8192, 256, 0, stream>>>(y1, ln2_g, ln2_b, y2);
  k_mo<<<dim3(8, 32), 256, 0, stream>>>(y2, mowb, mo_b, outp);
}
